// StateExtr_54348516164103
// MI455X (gfx1250) — matrix operand order repaired and checked by reading, not compiled, not run on hardware
//
#include <hip/hip_runtime.h>
#include <hip/hip_bf16.h>

// ---------------------------------------------------------------------------
// LayerNorm-LSTM (2 layers, B=128, T=1024, I=64, H=256) for gfx1250 / MI455X.
// Strategy:
//   * repack_kernel: f32 weights -> f16 WMMA B-fragment layout in d_ws,
//     concatenated [Wi | Wh] along K, with the unused i_gc block zeroed.
//   * lstm_kernel: 8 persistent workgroups x 512 threads (16 waves).
//     Each WG owns 16 batch rows for the full T loop. Per step:
//       gates[16,1024] = [x|h] (f16, LDS) x Wcat (f16, streamed from L2) via
//       v_wmma_f32_16x16x32_f16 (B fragments double-buffered over K);
//       then wave w does the 5 layernorms + sigmoid/tanh gate math for batch
//       row w in fp32 (c state in VGPRs), writes h back to LDS as f16, and
//       emits the final-LN masked output.
//   * Weight pointers are laundered through inline asm each call (prevents
//     LICM hoisting + scratch spilling across the time loop) and then cast
//     back to address_space(1) so loads lower as global_load_b128 (LOADcnt
//     only, saddr form) instead of flat_load (which also ties up DScnt).
// ---------------------------------------------------------------------------

typedef __attribute__((ext_vector_type(16))) _Float16 v16h;
typedef __attribute__((ext_vector_type(8)))  float    v8f;
typedef __attribute__((ext_vector_type(4)))  float    v4f;

// global (addrspace 1) views for the laundered weight pointer
typedef __attribute__((address_space(1))) const _Float16 GHalf;
typedef __attribute__((address_space(1))) const v16h     GV16h;

#define BB   128
#define TT   1024
#define II   64
#define HH   256
#define NTOT 1024              // 4*H
#define NT   64                // N tiles of 16
#define KT0  10                // layer0: K = 64 + 256 = 320 -> 10 k-steps of 32
#define KT1  16                // layer1: K = 256 + 256 = 512 -> 16 k-steps
#define GSTR 1028              // gates LDS row stride (floats), bank-staggered
#define HSTR 272               // lds_h row stride (halves), 16B-aligned, staggered
#define XSTR 72                // lds_x row stride (halves)

union AFrag { v16h v; v4f q[2]; };

// ---------------------------------------------------------------------------
// Weight repack: fragment index layout
//   frag(nt, kt) occupies 512 halves: half index = lane*16 + h
//   B-matrix (K x N) per-lane mapping:  n = nt*16 + (lane&15)
//                                       k = kt*32 + (lane>>4)*16 + h
// Layer 0 concat K: [0,64) = Wi0, [64,320) = Wh0.  Wi block zeroed for
// n in [512,768) (reference never uses i_gc from the input matmul).
// Layer 1 concat K: [0,256) = Wi1, [256,512) = Wh1, same zeroing.
// ---------------------------------------------------------------------------
__global__ void LNLSTM_repack_kernel(const float* __restrict__ Wi0,
                                     const float* __restrict__ Wh0,
                                     const float* __restrict__ Wi1,
                                     const float* __restrict__ Wh1,
                                     _Float16* __restrict__ w0,
                                     _Float16* __restrict__ w1) {
  const long n0 = (long)NT * KT0 * 512;
  const long n1 = (long)NT * KT1 * 512;
  for (long idx = (long)blockIdx.x * blockDim.x + threadIdx.x; idx < n0 + n1;
       idx += (long)gridDim.x * blockDim.x) {
    if (idx < n0) {
      const int h    = (int)(idx & 15);
      const int lane = (int)((idx >> 4) & 31);
      const long rem = idx >> 9;
      const int kt   = (int)(rem % KT0);
      const int nt   = (int)(rem / KT0);
      const int n = nt * 16 + (lane & 15);
      const int k = kt * 32 + ((lane >> 4) * 16) + h;
      float v;
      if (k < II) {
        v = (n >= 512 && n < 768) ? 0.f : Wi0[(long)n * II + k];
      } else {
        v = Wh0[(long)n * HH + (k - II)];
      }
      w0[idx] = (_Float16)v;
    } else {
      const long i1  = idx - n0;
      const int h    = (int)(i1 & 15);
      const int lane = (int)((i1 >> 4) & 31);
      const long rem = i1 >> 9;
      const int kt   = (int)(rem % KT1);
      const int nt   = (int)(rem / KT1);
      const int n = nt * 16 + (lane & 15);
      const int k = kt * 32 + ((lane >> 4) * 16) + h;
      float v;
      if (k < HH) {
        v = (n >= 512 && n < 768) ? 0.f : Wi1[(long)n * HH + k];
      } else {
        v = Wh1[(long)n * HH + (k - HH)];
      }
      w1[i1] = (_Float16)v;
    }
  }
}

// ---------------------------------------------------------------------------
// Fused [A0|A1] x Wcat matmul for one layer.  Each wave computes 4 N-tiles,
// B fragments double-buffered one k-step ahead so global_load latency hides
// behind the 4 WMMAs of the current k-step.
//   A fragment (16x32 f16, contiguous k as in repack): lane -> m = lane&15,
//     halves 0..15 = k (kt*32 + ksel .. +15),
//     ksel = (lane>=16) ? 16 : 0  -> two 16B LDS loads per fragment.
// ---------------------------------------------------------------------------
template <int KT, int LIM0>
__device__ __forceinline__ void mm_layer(const _Float16* __restrict__ wp_in,
                                         const _Float16* __restrict__ a0, int s0,
                                         const _Float16* __restrict__ a1, int s1,
                                         float* __restrict__ gates,
                                         int wave, int lane) {
  // Launder the weight pointer: opaque per call, so the weight loads cannot
  // be hoisted/CSE'd across the outer time loop (avoids scratch spilling).
  // Rebuild it as an address_space(1) pointer so loads stay on the GLOBAL
  // path (global_load_b128, LOADcnt only) instead of degrading to FLAT.
  unsigned long long wpu = (unsigned long long)wp_in;
  asm volatile("" : "+s"(wpu));
  const GHalf* wp = (const GHalf*)wpu;

  const int mrow  = lane & 15;
  const int khalf = (lane >> 4) * 16;
  const _Float16* __restrict__ r0 = a0 + mrow * s0;
  const _Float16* __restrict__ r1 = a1 + mrow * s1;

  const v8f z = {0.f, 0.f, 0.f, 0.f, 0.f, 0.f, 0.f, 0.f};
  v8f acc0 = z, acc1 = z, acc2 = z, acc3 = z;

  const int nt0 = wave * 4;
  const size_t nts = (size_t)KT * 512;                  // fragment stride per nt
  const GHalf* bp = wp + (size_t)nt0 * nts + lane * 16;

  // prologue: load k-step 0
  v16h bc0 = *(const GV16h*)(bp + 0 * nts);
  v16h bc1 = *(const GV16h*)(bp + 1 * nts);
  v16h bc2 = *(const GV16h*)(bp + 2 * nts);
  v16h bc3 = *(const GV16h*)(bp + 3 * nts);

#pragma unroll
  for (int kt = 0; kt < KT; ++kt) {
    v16h bn0, bn1, bn2, bn3;
    if (kt + 1 < KT) {
      const GHalf* bpn = bp + (size_t)(kt + 1) * 512;
      bn0 = *(const GV16h*)(bpn + 0 * nts);
      bn1 = *(const GV16h*)(bpn + 1 * nts);
      bn2 = *(const GV16h*)(bpn + 2 * nts);
      bn3 = *(const GV16h*)(bpn + 3 * nts);
    }

    const int kg = kt * 32;
    const _Float16* row = (kg < LIM0) ? (r0 + kg) : (r1 + (kg - LIM0));
    AFrag a;
    a.q[0] = *(const v4f*)(row + khalf);
    a.q[1] = *(const v4f*)(row + khalf + 8);

    acc0 = __builtin_amdgcn_wmma_f32_16x16x32_f16(false, a.v, false, bc0,
                                                  (short)0, acc0, false, false);
    acc1 = __builtin_amdgcn_wmma_f32_16x16x32_f16(false, a.v, false, bc1,
                                                  (short)0, acc1, false, false);
    acc2 = __builtin_amdgcn_wmma_f32_16x16x32_f16(false, a.v, false, bc2,
                                                  (short)0, acc2, false, false);
    acc3 = __builtin_amdgcn_wmma_f32_16x16x32_f16(false, a.v, false, bc3,
                                                  (short)0, acc3, false, false);

    if (kt + 1 < KT) { bc0 = bn0; bc1 = bn1; bc2 = bn2; bc3 = bn3; }
  }

  // C/D layout: VGPR r -> M = r (lanes 0-15) or M = r+8 (lanes 16-31),
  // N = lane&15
  const int rbase = (lane >> 4) << 3;
  const int cb    = lane & 15;
#pragma unroll
  for (int r = 0; r < 8; ++r) {
    gates[(size_t)(rbase + r) * GSTR + (nt0 + 0) * 16 + cb] = acc0[r];
    gates[(size_t)(rbase + r) * GSTR + (nt0 + 1) * 16 + cb] = acc1[r];
    gates[(size_t)(rbase + r) * GSTR + (nt0 + 2) * 16 + cb] = acc2[r];
    gates[(size_t)(rbase + r) * GSTR + (nt0 + 3) * 16 + cb] = acc3[r];
  }
}

__device__ __forceinline__ float wsum32(float v) {
#pragma unroll
  for (int m = 16; m; m >>= 1) v += __shfl_xor(v, m, 32);
  return v;
}

// LayerNorm over 256 elements held as 8 per lane (col = lane + 32*j).
__device__ __forceinline__ void ln256(float (&v)[8], const float* __restrict__ g,
                                      const float* __restrict__ b, int lane) {
  float s = 0.f, s2 = 0.f;
#pragma unroll
  for (int j = 0; j < 8; ++j) { s += v[j]; s2 += v[j] * v[j]; }
  s  = wsum32(s);
  s2 = wsum32(s2);
  const float mu  = s * (1.f / 256.f);
  const float var = s2 * (1.f / 256.f) - mu * mu;
  const float inv = rsqrtf(var + 1e-5f);
#pragma unroll
  for (int j = 0; j < 8; ++j) {
    const int c = lane + 32 * j;
    v[j] = (v[j] - mu) * inv * g[c] + b[c];
  }
}

__device__ __forceinline__ float sigm(float x) { return 1.f / (1.f + __expf(-x)); }

// Gate math for one batch row (one wave): consumes gates row, updates cell
// state registers, returns new h (fp32, 8 per lane).
__device__ __forceinline__ void cell_gates(const float* __restrict__ grow,
                                           const float* __restrict__ lg,
                                           const float* __restrict__ lb,
                                           float (&cr)[8], float (&h)[8],
                                           int lane) {
  float zi[8], zf[8], hg[8], zo[8];
#pragma unroll
  for (int j = 0; j < 8; ++j) {
    const int c = lane + 32 * j;
    zi[j] = grow[c];
    zf[j] = grow[256 + c];
    hg[j] = grow[512 + c];   // chunk 2 = h_g only (Wi block zeroed in repack)
    zo[j] = grow[768 + c];
  }
  ln256(zi, lg + 0 * HH, lb + 0 * HH, lane);
#pragma unroll
  for (int j = 0; j < 8; ++j) zi[j] = sigm(zi[j]);          // i gate
  ln256(zf, lg + 1 * HH, lb + 1 * HH, lane);
#pragma unroll
  for (int j = 0; j < 8; ++j) zf[j] = sigm(zf[j]);          // f gate
  float zg[8];
#pragma unroll
  for (int j = 0; j < 8; ++j) zg[j] = zi[j] + hg[j];        // sigmoid(i) + h_g
  ln256(zg, lg + 2 * HH, lb + 2 * HH, lane);
#pragma unroll
  for (int j = 0; j < 8; ++j) zg[j] = tanhf(zg[j]);         // candidate
  ln256(zo, lg + 3 * HH, lb + 3 * HH, lane);
#pragma unroll
  for (int j = 0; j < 8; ++j) zo[j] = sigm(zo[j]);          // o gate
  float cn[8];
#pragma unroll
  for (int j = 0; j < 8; ++j) cn[j] = zf[j] * cr[j] + zi[j] * zg[j];
  ln256(cn, lg + 4 * HH, lb + 4 * HH, lane);                // c = LN(...)
#pragma unroll
  for (int j = 0; j < 8; ++j) {
    cr[j] = cn[j];
    h[j]  = zo[j] * tanhf(cn[j]);
  }
}

__global__ __launch_bounds__(512, 1) void LNLSTM_main_kernel(
    const float* __restrict__ x, const int* __restrict__ seq_len,
    const _Float16* __restrict__ w0, const _Float16* __restrict__ w1,
    const float* __restrict__ ln_gamma, const float* __restrict__ ln_beta,
    const float* __restrict__ final_gamma, const float* __restrict__ final_beta,
    float* __restrict__ out) {
  __shared__ float    lds_gates[16 * GSTR];          // 65,792 B
  __shared__ _Float16 lds_h[2][16][HSTR];            // 17,408 B
  __shared__ _Float16 lds_x[16][XSTR];               //  2,304 B

  const int tid  = threadIdx.x;
  const int wave = tid >> 5;    // 0..15 == local batch row
  const int lane = tid & 31;
  const int bg   = blockIdx.x * 16 + wave;           // global batch row (duty)

  // zero recurrent state
  for (int i = tid; i < 2 * 16 * HSTR; i += 512)
    ((_Float16*)lds_h)[i] = (_Float16)0.f;
  float c0r[8], c1r[8], h[8];
#pragma unroll
  for (int j = 0; j < 8; ++j) { c0r[j] = 0.f; c1r[j] = 0.f; }

  const int slen = seq_len[bg];
  const float* lg0 = ln_gamma;                // [2][5][256]
  const float* lb0 = ln_beta;
  const float* lg1 = ln_gamma + 5 * HH;
  const float* lb1 = ln_beta + 5 * HH;

  _Float16* h0p = &lds_h[0][0][0];
  _Float16* h1p = &lds_h[1][0][0];
  _Float16* xp  = &lds_x[0][0];

  __syncthreads();

  for (int t = 0; t < TT; ++t) {
    // stage x_t (64 f32 -> f16) for this wave's row
    const float* xr = x + ((size_t)bg * TT + t) * II;
    lds_x[wave][lane]      = (_Float16)xr[lane];
    lds_x[wave][lane + 32] = (_Float16)xr[lane + 32];
    __syncthreads();

    // layer 0: [x | h0_prev] x W0cat  (K = 64 + 256)
    mm_layer<KT0, II>(w0, xp, XSTR, h0p, HSTR, lds_gates, wave, lane);
    __syncthreads();

    cell_gates(&lds_gates[(size_t)wave * GSTR], lg0, lb0, c0r, h, lane);
#pragma unroll
    for (int j = 0; j < 8; ++j)
      lds_h[0][wave][lane + 32 * j] = (_Float16)h[j];
    __syncthreads();

    // layer 1: [h0_new | h1_prev] x W1cat  (K = 256 + 256)
    mm_layer<KT1, HH>(w1, h0p, HSTR, h1p, HSTR, lds_gates, wave, lane);
    __syncthreads();

    cell_gates(&lds_gates[(size_t)wave * GSTR], lg1, lb1, c1r, h, lane);
#pragma unroll
    for (int j = 0; j < 8; ++j)
      lds_h[1][wave][lane + 32 * j] = (_Float16)h[j];

    // final output: LN(h1 * mask); masked rows -> LN(0) == final_beta
    float* orow = out + ((size_t)bg * TT + t) * HH;
    if (t < slen) {
      float o[8];
#pragma unroll
      for (int j = 0; j < 8; ++j) o[j] = h[j];
      ln256(o, final_gamma, final_beta, lane);
#pragma unroll
      for (int j = 0; j < 8; ++j) orow[lane + 32 * j] = o[j];
    } else {
#pragma unroll
      for (int j = 0; j < 8; ++j) {
        const int c = lane + 32 * j;
        orow[c] = final_beta[c];
      }
    }
    __syncthreads();
  }
}

extern "C" void kernel_launch(void* const* d_in, const int* in_sizes, int n_in,
                              void* d_out, int out_size, void* d_ws, size_t ws_size,
                              hipStream_t stream) {
  (void)in_sizes; (void)n_in; (void)out_size; (void)ws_size;
  const float* x    = (const float*)d_in[0];
  const int*   seq  = (const int*)d_in[1];
  const float* Wi0  = (const float*)d_in[2];
  const float* Wh0  = (const float*)d_in[3];
  const float* Wi1  = (const float*)d_in[4];
  const float* Wh1  = (const float*)d_in[5];
  const float* lg   = (const float*)d_in[6];
  const float* lb   = (const float*)d_in[7];
  const float* fg   = (const float*)d_in[8];
  const float* fb   = (const float*)d_in[9];
  float* out = (float*)d_out;

  _Float16* w0 = (_Float16*)d_ws;                       // 64*10*512 halves
  _Float16* w1 = w0 + (size_t)NT * KT0 * 512;           // 64*16*512 halves

  LNLSTM_repack_kernel<<<1024, 256, 0, stream>>>(Wi0, Wh0, Wi1, Wh1, w0, w1);
  LNLSTM_main_kernel<<<BB / 16, 512, 0, stream>>>(x, seq, w0, w1, lg, lb, fg, fb,
                                                  out);
}
